// GIN_27041114096324
// MI455X (gfx1250) — matrix operand order repaired and checked by reading, not compiled, not run on hardware
//
#include <hip/hip_runtime.h>

// ---------------------------------------------------------------------------
// Types for CDNA5 WMMA (gfx1250, wave32)
// ---------------------------------------------------------------------------
typedef __bf16 bf16_t;
typedef __attribute__((ext_vector_type(16))) __bf16 bf16x16;
typedef __attribute__((ext_vector_type(8)))  __bf16 bf16x8;
typedef __attribute__((ext_vector_type(8)))  float  f32x8;

union ABFrag { bf16x16 v; bf16x8 h[2]; };

#define N_NODES 50000
#define N_EDGES 800000
#define F_IN    96
#define HID     128
#define NCLS    40
#define NCLS_PAD 48

// ---------------------------------------------------------------------------
// Elementwise helpers
// ---------------------------------------------------------------------------
__global__ void gin_zero_f32(float* __restrict__ p, int n) {
    int i = blockIdx.x * blockDim.x + threadIdx.x;
    if (i < n) p[i] = 0.0f;
}

// o = bf16(a + b)
__global__ void gin_add_cvt(const float* __restrict__ a, const float* __restrict__ b,
                            bf16_t* __restrict__ o, int n) {
    int i = blockIdx.x * blockDim.x + threadIdx.x;
    if (i < n) o[i] = (bf16_t)(a[i] + b[i]);
}

// Wt[col][k] = bf16(W[k][col]), zero-padding cols in [ncols, npad)
__global__ void gin_wcvt_t(const float* __restrict__ W, bf16_t* __restrict__ Wt,
                           int K, int ncols, int npad) {
    int i = blockIdx.x * blockDim.x + threadIdx.x;
    int total = npad * K;
    if (i >= total) return;
    int col = i / K;
    int k   = i - col * K;
    float v = (col < ncols) ? W[(size_t)k * ncols + col] : 0.0f;
    Wt[(size_t)col * K + k] = (bf16_t)v;
}

// ---------------------------------------------------------------------------
// Scatter-add aggregation: agg[dst[e]] += x[src[e]]  (F multiple of 4)
// One thread handles 4 features of one edge: b128 gather + 4 f32 atomics.
// x/agg are L2-resident (<=26MB vs 192MB L2) so this is L2 atomic traffic.
// ---------------------------------------------------------------------------
__global__ void gin_scatter_add(const float* __restrict__ x,
                                const int* __restrict__ src,
                                const int* __restrict__ dst,
                                float* __restrict__ agg,
                                int nE, int F) {
    int tid = blockIdx.x * blockDim.x + threadIdx.x;
    int chunks = F >> 2;
    int total = nE * chunks;
    if (tid >= total) return;
    int e = tid / chunks;
    int c = tid - e * chunks;
    int s = src[e];
    int d = dst[e];
    const float4 v = *(const float4*)(x + (size_t)s * F + (size_t)c * 4);
    float* p = agg + (size_t)d * F + (size_t)c * 4;
    atomicAdd(p + 0, v.x);
    atomicAdd(p + 1, v.y);
    atomicAdd(p + 2, v.z);
    atomicAdd(p + 3, v.w);
}

// ---------------------------------------------------------------------------
// WMMA GEMM: C[M, ncols] = act(A[M,K] @ Wt[npad,K]^T + bias)
//   A  : bf16 row-major [M][K], K % 32 == 0, M % 16 == 0
//   Wt : bf16 row-major [npad][K]  (i.e. weights transposed; npad % 16 == 0)
//   One wave per 16x16 tile; blockDim = (32, npad/16); grid = M/16.
//   v_wmma_f32_16x16x32_bf16 accumulate chain over K.
// Fragment layouts used here (wave32, contiguous k order, not the ISA table order):
//   A 16x32 bf16: lane<16 -> K[k0+0..15]; lane>=16 -> K[k0+16..31]
//   B 32x16 bf16: lane = column; lane<16 -> K[k0+0..15]; lane>=16 -> K[k0+16..31]
//   C/D: lane = column (mod 16); VGPR r -> row r (lane<16) / row r+8 (lane>=16)
// ---------------------------------------------------------------------------
template<bool RELU, bool OUT_BF16>
__global__ __launch_bounds__(256)
void gin_wmma_gemm(const bf16_t* __restrict__ A,
                   const bf16_t* __restrict__ Wt,
                   const float*  __restrict__ bias,
                   float*  __restrict__ Cf,
                   bf16_t* __restrict__ Cb,
                   int K, int ncols, int ldc) {
    const int lane    = threadIdx.x;        // 0..31
    const int ntile   = threadIdx.y;        // column tile index
    const int row0    = blockIdx.x * 16;    // M tile
    const int halfsel = lane >> 4;          // 0 | 1
    const int l15     = lane & 15;

    const bf16_t* __restrict__ arow = A  + (size_t)(row0 + l15) * K;
    const bf16_t* __restrict__ bcol = Wt + (size_t)(ntile * 16 + l15) * K;
    const int akb = halfsel * 16;           // A lane k-base within 32-chunk
    const int bkb = halfsel * 16;           // B lane k-base within 32-chunk

    f32x8 acc = {};
    for (int k0 = 0; k0 < K; k0 += 32) {
        ABFrag a, b;
        a.h[0] = *(const bf16x8*)(arow + k0 + akb);
        a.h[1] = *(const bf16x8*)(arow + k0 + akb + 8);
        b.h[0] = *(const bf16x8*)(bcol + k0 + bkb);
        b.h[1] = *(const bf16x8*)(bcol + k0 + bkb + 8);
        acc = __builtin_amdgcn_wmma_f32_16x16x32_bf16(
                  false, a.v, false, b.v, (short)0, acc, false, false);
    }

    const int col = ntile * 16 + l15;
    if (col < ncols) {                       // after all WMMAs: EXEC all-1s above
        const float bv = bias[col];
        #pragma unroll
        for (int r = 0; r < 8; ++r) {
            const int row = row0 + r + halfsel * 8;
            float v = acc[r] + bv;
            if (RELU) v = v > 0.0f ? v : 0.0f;
            if (OUT_BF16) Cb[(size_t)row * ldc + col] = (bf16_t)v;
            else          Cf[(size_t)row * ldc + col] = v;
        }
    }
}

// ---------------------------------------------------------------------------
// Host-side launch
// ---------------------------------------------------------------------------
static inline int cdiv(int a, int b) { return (a + b - 1) / b; }

extern "C" void kernel_launch(void* const* d_in, const int* in_sizes, int n_in,
                              void* d_out, int out_size, void* d_ws, size_t ws_size,
                              hipStream_t stream) {
    (void)in_sizes; (void)n_in; (void)out_size; (void)ws_size;

    const float* x    = (const float*)d_in[0];
    const int*   eidx = (const int*)d_in[1];       // [2, E]; JAX demotes int64->int32
    const float* W1a  = (const float*)d_in[2];
    const float* b1a  = (const float*)d_in[3];
    const float* W2a  = (const float*)d_in[4];
    const float* b2a  = (const float*)d_in[5];
    const float* W1b  = (const float*)d_in[6];
    const float* b1b  = (const float*)d_in[7];
    const float* W2b  = (const float*)d_in[8];
    const float* b2b  = (const float*)d_in[9];
    const float* Wfc  = (const float*)d_in[10];
    const float* bfc  = (const float*)d_in[11];
    float* out = (float*)d_out;

    const int* srcI = eidx;
    const int* dstI = eidx + N_EDGES;

    // ---- workspace layout ----
    char* ws = (char*)d_ws;
    float*  agg  = (float*)ws;                                  // N*128 f32
    float*  hbuf = agg + (size_t)N_NODES * HID;                 // N*128 f32
    bf16_t* zbf  = (bf16_t*)(hbuf + (size_t)N_NODES * HID);     // N*128 bf16
    bf16_t* gA   = zbf + (size_t)N_NODES * HID;                 // N*128 bf16
    bf16_t* gB   = gA  + (size_t)N_NODES * HID;                 // N*128 bf16
    bf16_t* w1aT = gB  + (size_t)N_NODES * HID;                 // [128][96]
    bf16_t* w2aT = w1aT + (size_t)HID * F_IN;                   // [128][128]
    bf16_t* w1bT = w2aT + (size_t)HID * HID;
    bf16_t* w2bT = w1bT + (size_t)HID * HID;
    bf16_t* wfcT = w2bT + (size_t)HID * HID;                    // [48][128]

    const int TPB = 256;
    const int mtiles = N_NODES / 16;                            // 3125

    // ---- weight convert + transpose (bf16) ----
    gin_wcvt_t<<<cdiv(HID * F_IN, TPB), TPB, 0, stream>>>(W1a, w1aT, F_IN, HID, HID);
    gin_wcvt_t<<<cdiv(HID * HID,  TPB), TPB, 0, stream>>>(W2a, w2aT, HID, HID, HID);
    gin_wcvt_t<<<cdiv(HID * HID,  TPB), TPB, 0, stream>>>(W1b, w1bT, HID, HID, HID);
    gin_wcvt_t<<<cdiv(HID * HID,  TPB), TPB, 0, stream>>>(W2b, w2bT, HID, HID, HID);
    gin_wcvt_t<<<cdiv(NCLS_PAD * HID, TPB), TPB, 0, stream>>>(Wfc, wfcT, HID, NCLS, NCLS_PAD);

    // ================= Layer 1 =================
    {   // agg1 = scatter_add(x[src] -> dst), F = 96
        const int n = N_NODES * F_IN;
        gin_zero_f32<<<cdiv(n, TPB), TPB, 0, stream>>>(agg, n);
        const int work = N_EDGES * (F_IN / 4);
        gin_scatter_add<<<cdiv(work, TPB), TPB, 0, stream>>>(x, srcI, dstI, agg, N_EDGES, F_IN);
        // z1 = bf16(x + agg1)
        gin_add_cvt<<<cdiv(n, TPB), TPB, 0, stream>>>(x, agg, zbf, n);
    }
    // gA = relu(z1 @ W1a + b1a)            [N,128] bf16
    gin_wmma_gemm<true, true><<<mtiles, dim3(32, HID / 16), 0, stream>>>(
        zbf, w1aT, b1a, nullptr, gA, F_IN, HID, HID);
    // h = relu(gA @ W2a + b2a)             [N,128] f32   (outer relu fused)
    gin_wmma_gemm<true, false><<<mtiles, dim3(32, HID / 16), 0, stream>>>(
        gA, w2aT, b2a, hbuf, nullptr, HID, HID, HID);

    // ================= Layer 2 =================
    {   // agg2 = scatter_add(h[src] -> dst), F = 128
        const int n = N_NODES * HID;
        gin_zero_f32<<<cdiv(n, TPB), TPB, 0, stream>>>(agg, n);
        const int work = N_EDGES * (HID / 4);
        gin_scatter_add<<<cdiv(work, TPB), TPB, 0, stream>>>(hbuf, srcI, dstI, agg, N_EDGES, HID);
        // z2 = bf16(h + agg2)
        gin_add_cvt<<<cdiv(n, TPB), TPB, 0, stream>>>(hbuf, agg, zbf, n);
    }
    // gA = relu(z2 @ W1b + b1b)            [N,128] bf16
    gin_wmma_gemm<true, true><<<mtiles, dim3(32, HID / 16), 0, stream>>>(
        zbf, w1bT, b1b, nullptr, gA, HID, HID, HID);
    // gB = gA @ W2b + b2b                  [N,128] bf16 (no relu before FC)
    gin_wmma_gemm<false, true><<<mtiles, dim3(32, HID / 16), 0, stream>>>(
        gA, w2bT, b2b, nullptr, gB, HID, HID, HID);

    // ================= FC head =================
    // out = gB @ Wfc + bfc                 [N,40] f32 (padded to 48 cols, masked store)
    gin_wmma_gemm<false, false><<<mtiles, dim3(32, NCLS_PAD / 16), 0, stream>>>(
        gB, wfcT, bfc, out, nullptr, HID, NCLS, NCLS);
}
